// FastSpeech2Pros_45655502357069
// MI455X (gfx1250) — compile-verified, matrix operand order checked by reading, not run on hardware
//
#include <hip/hip_runtime.h>
#include <hip/hip_bf16.h>

#define B_    16
#define T_    512
#define MM_   4096
#define D_    256
#define MEL_  80
#define MELP_ 96
#define ROWS_ (B_ * MM_)   // 65536
#define MTOK_ (B_ * T_)    // 8192

typedef __attribute__((ext_vector_type(16))) __bf16 v16bf;
typedef __attribute__((ext_vector_type(8)))  float  v8f;

__device__ __forceinline__ __bf16 f2bf(float f) {
  unsigned u = __builtin_bit_cast(unsigned, f);
  unsigned r = (u + 0x7FFFu + ((u >> 16) & 1u)) >> 16;
  return __builtin_bit_cast(__bf16, (unsigned short)r);
}
__device__ __forceinline__ __bf16 bf0() {
  return __builtin_bit_cast(__bf16, (unsigned short)0);
}

union TileU { uint4 q[2]; v16bf v; };

// A/B tile (16 rows x 32 K) loader in CDNA5 wave32 WMMA layout:
// lane L: row = L&15, half = L>>4 selects K ranges {h*8..h*8+7, 16+h*8..16+h*8+7}
__device__ __forceinline__ v16bf load_tile(const __bf16* base, int ld) {
  int l = threadIdx.x & 31;
  const __bf16* p = base + (size_t)(l & 15) * ld + ((l >> 4) << 3);
  TileU t;
  t.q[0] = *(const uint4*)(p);
  t.q[1] = *(const uint4*)(p + 16);
  return t.v;
}

// Per-lane row pointer (already includes k0); used for gathers / shifted conv rows.
__device__ __forceinline__ v16bf load_tile_row(const __bf16* rowp, bool valid) {
  int h = (threadIdx.x & 31) >> 4;
  TileU t;
  if (valid) {
    const __bf16* p = rowp + h * 8;
    t.q[0] = *(const uint4*)(p);
    t.q[1] = *(const uint4*)(p + 16);
  } else {
    t.q[0] = make_uint4(0, 0, 0, 0);
    t.q[1] = make_uint4(0, 0, 0, 0);
  }
  return t.v;
}

__device__ __forceinline__ v8f wmma_bf16(v16bf a, v16bf b, v8f c) {
  return __builtin_amdgcn_wmma_f32_16x16x32_bf16(false, a, false, b, (short)0, c, false, false);
}

// ---------------- weight prep ----------------

// dst[n*Kp + k] = (k < K) ? bf16(src[k*N + n]) : 0   (transpose + pad K)
__global__ void k_transpose_bf16(const float* __restrict__ src, __bf16* __restrict__ dst,
                                 int K, int N, int Kp) {
  int i = blockIdx.x * 256 + threadIdx.x;
  if (i >= N * Kp) return;
  int n = i / Kp, k = i - n * Kp;
  dst[i] = (k < K) ? f2bf(src[(size_t)k * N + n]) : bf0();
}

__global__ void k_convert_bf16(const float* __restrict__ src, __bf16* __restrict__ dst, int total) {
  int i = blockIdx.x * 256 + threadIdx.x;
  if (i < total) dst[i] = f2bf(src[i]);
}

__global__ void k_zero_bf16(__bf16* __restrict__ dst, int total) {
  int i = blockIdx.x * 256 + threadIdx.x;
  if (i < total) dst[i] = bf0();
}

// ---------------- encoder: enc = tanh(emb[texts]@enc_w + b)*mask ----------------

__global__ void __launch_bounds__(256) k_encoder(
    const int* __restrict__ texts, const int* __restrict__ src_lens,
    const float* __restrict__ spk, const __bf16* __restrict__ embB,
    const __bf16* __restrict__ encWT, const float* __restrict__ enc_b,
    __bf16* __restrict__ h_si_bf, __bf16* __restrict__ h_sd_bf,
    float* __restrict__ h_sd_f) {
  int wave = (blockIdx.x * blockDim.x + threadIdx.x) >> 5;
  const int nTiles = D_ / 16;
  int mTile = wave / nTiles, nTile = wave - mTile * nTiles;
  if (mTile * 16 >= MTOK_) return;
  int l = threadIdx.x & 31, lm = l & 15, h = l >> 4;
  int n0 = nTile * 16, nn = n0 + lm;
  v8f acc;
#pragma unroll
  for (int r = 0; r < 8; ++r) acc[r] = enc_b[nn];
  int gm = mTile * 16 + lm;
  const __bf16* arow = embB + (size_t)texts[gm] * D_;
#pragma unroll
  for (int kc = 0; kc < D_ / 32; ++kc) {
    v16bf a = load_tile_row(arow + kc * 32, true);
    v16bf b = load_tile(encWT + (size_t)n0 * D_ + kc * 32, D_);
    acc = wmma_bf16(a, b, acc);
  }
#pragma unroll
  for (int r = 0; r < 8; ++r) {
    int m = mTile * 16 + r + 8 * h;
    int bI = m >> 9, t = m & (T_ - 1);
    float e = tanhf(acc[r]);
    if (t >= src_lens[bI]) e = 0.f;
    float hs = e + spk[(size_t)bI * D_ + nn];
    size_t o = (size_t)m * D_ + nn;
    h_si_bf[o] = f2bf(e);
    h_sd_bf[o] = f2bf(hs);
    h_sd_f[o]  = hs;
  }
}

// ---------------- pre = h_sd@W_a + h_si@W_b + b1 (parallel part of scan) ----------------

__global__ void __launch_bounds__(256) k_pre(
    const __bf16* __restrict__ h_sd_bf, const __bf16* __restrict__ h_si_bf,
    const __bf16* __restrict__ WaT, const __bf16* __restrict__ WbT,
    const float* __restrict__ b1, float* __restrict__ pre) {
  int wave = (blockIdx.x * blockDim.x + threadIdx.x) >> 5;
  const int nTiles = D_ / 16;
  int mTile = wave / nTiles, nTile = wave - mTile * nTiles;
  if (mTile * 16 >= MTOK_) return;
  int l = threadIdx.x & 31, lm = l & 15, h = l >> 4;
  int n0 = nTile * 16, nn = n0 + lm;
  v8f acc;
#pragma unroll
  for (int r = 0; r < 8; ++r) acc[r] = b1[nn];
  const __bf16* A0 = h_sd_bf + (size_t)mTile * 16 * D_;
  const __bf16* A1 = h_si_bf + (size_t)mTile * 16 * D_;
#pragma unroll
  for (int kc = 0; kc < D_ / 32; ++kc)
    acc = wmma_bf16(load_tile(A0 + kc * 32, D_), load_tile(WaT + (size_t)n0 * D_ + kc * 32, D_), acc);
#pragma unroll
  for (int kc = 0; kc < D_ / 32; ++kc)
    acc = wmma_bf16(load_tile(A1 + kc * 32, D_), load_tile(WbT + (size_t)n0 * D_ + kc * 32, D_), acc);
#pragma unroll
  for (int r = 0; r < 8; ++r) {
    int m = mTile * 16 + r + 8 * h;
    pre[(size_t)m * D_ + nn] = acc[r];
  }
}

// ---------------- sequential prenet scan: one persistent WG, weights in LDS ----------------
// Weight preload uses gfx1250 async global->LDS DMA (ASYNCcnt path).

__global__ void __launch_bounds__(512) k_scan(
    const __bf16* __restrict__ WcT_g, const __bf16* __restrict__ W2T_g,
    const float* __restrict__ pre, const float* __restrict__ b2,
    const float* __restrict__ h_sd_f, __bf16* __restrict__ outrow_bf) {
  extern __shared__ __bf16 smem[];
  __bf16* WcT   = smem;                       // 256*256 @ LDS byte 0
  __bf16* W2T   = smem + D_ * D_;             // 256*256 @ LDS byte 131072
  __bf16* prevb = smem + 2 * D_ * D_;         // 16*256
  __bf16* hb    = prevb + B_ * D_;            // 16*256

  // async copy 2 x 128KB weight panels into LDS (16B per lane per issue)
  for (int i = threadIdx.x * 8; i < D_ * D_; i += 512 * 8) {
    unsigned off0 = (unsigned)(i * 2);
    unsigned off1 = (unsigned)(D_ * D_ * 2 + i * 2);
    asm volatile("global_load_async_to_lds_b128 %0, %1, off"
                 :: "v"(off0), "v"(WcT_g + i) : "memory");
    asm volatile("global_load_async_to_lds_b128 %0, %1, off"
                 :: "v"(off1), "v"(W2T_g + i) : "memory");
  }
  asm volatile("s_wait_asynccnt 0x0" ::: "memory");
  for (int i = threadIdx.x; i < B_ * D_; i += 512) prevb[i] = bf0();
  __syncthreads();

  int wave = threadIdx.x >> 5;
  int l = threadIdx.x & 31, lm = l & 15, h = l >> 4;
  int n0 = wave * 16, nn = n0 + lm;

  for (int t = 0; t < T_; ++t) {
    v8f acc;
#pragma unroll
    for (int r = 0; r < 8; ++r) {
      int m = r + 8 * h;                      // batch index
      acc[r] = pre[((size_t)m * T_ + t) * D_ + nn];
    }
#pragma unroll
    for (int kc = 0; kc < D_ / 32; ++kc)
      acc = wmma_bf16(load_tile(prevb + kc * 32, D_), load_tile(WcT + (size_t)n0 * D_ + kc * 32, D_), acc);
#pragma unroll
    for (int r = 0; r < 8; ++r) {
      int m = r + 8 * h;
      hb[(size_t)m * D_ + nn] = f2bf(tanhf(acc[r]));
    }
    __syncthreads();
    v8f acc2;
#pragma unroll
    for (int r = 0; r < 8; ++r) acc2[r] = b2[nn];
#pragma unroll
    for (int kc = 0; kc < D_ / 32; ++kc)
      acc2 = wmma_bf16(load_tile(hb + kc * 32, D_), load_tile(W2T + (size_t)n0 * D_ + kc * 32, D_), acc2);
#pragma unroll
    for (int r = 0; r < 8; ++r) {
      int m = r + 8 * h;
      float o = acc2[r];
      prevb[(size_t)m * D_ + nn] = f2bf(o);
      size_t row = (size_t)m * T_ + t;
      outrow_bf[row * D_ + nn] = f2bf(o + h_sd_f[row * D_ + nn]);
    }
    __syncthreads();
  }
}

// ---------------- length regulator ----------------

__global__ void k_cumsum(const int* __restrict__ d_targets, const int* __restrict__ src_lens,
                         int* __restrict__ cum, int* __restrict__ total) {
  int b = threadIdx.x;
  if (b >= B_) return;
  int s = 0, L = src_lens[b];
  for (int t = 0; t < T_; ++t) {
    int dv = (t < L) ? d_targets[b * T_ + t] : 0;
    s += dv;
    cum[b * T_ + t] = s;
  }
  total[b] = s;
}

__global__ void k_index(const int* __restrict__ cum, int* __restrict__ idx) {
  int i = blockIdx.x * 256 + threadIdx.x;
  if (i >= B_ * MM_) return;
  int b = i >> 12, pos = i & (MM_ - 1);
  const int* c = cum + b * T_;
  int lo = 0, hi = T_;
  while (lo < hi) { int mid = (lo + hi) >> 1; if (c[mid] <= pos) lo = mid + 1; else hi = mid; }
  idx[i] = (lo > T_ - 1) ? (T_ - 1) : lo;
}

__global__ void k_expand(const __bf16* __restrict__ outrow_bf, const int* __restrict__ idx,
                         const int* __restrict__ total, __bf16* __restrict__ expb) {
  int i = blockIdx.x * 256 + threadIdx.x;
  if (i >= B_ * MM_ * D_) return;
  int row = i >> 8, n = i & (D_ - 1);
  int b = row >> 12, pos = row & (MM_ - 1);
  __bf16 v = bf0();
  if (pos < total[b]) v = outrow_bf[((size_t)b * T_ + idx[row]) * D_ + n];
  expb[i] = v;
}

// ---------------- dec = tanh(exp @ dec_w + b), 16x64 per wave (A reused 4x) ----------------

__global__ void __launch_bounds__(256) k_dec(
    const __bf16* __restrict__ expb, const __bf16* __restrict__ decWT,
    const float* __restrict__ dec_b, __bf16* __restrict__ decb) {
  int wave = (blockIdx.x * blockDim.x + threadIdx.x) >> 5;
  const int nGroups = D_ / 64; // 4
  int mTile = wave / nGroups, g = wave - mTile * nGroups;
  if (mTile * 16 >= ROWS_) return;
  int l = threadIdx.x & 31, lm = l & 15, h = l >> 4;
  int n0 = g * 64;
  v8f acc[4];
#pragma unroll
  for (int j = 0; j < 4; ++j)
#pragma unroll
    for (int r = 0; r < 8; ++r) acc[j][r] = dec_b[n0 + 16 * j + lm];
  const __bf16* A = expb + (size_t)mTile * 16 * D_;
#pragma unroll
  for (int kc = 0; kc < D_ / 32; ++kc) {
    v16bf a = load_tile(A + kc * 32, D_);
#pragma unroll
    for (int j = 0; j < 4; ++j)
      acc[j] = wmma_bf16(a, load_tile(decWT + (size_t)(n0 + 16 * j) * D_ + kc * 32, D_), acc[j]);
  }
#pragma unroll
  for (int j = 0; j < 4; ++j)
#pragma unroll
    for (int r = 0; r < 8; ++r) {
      int m = mTile * 16 + r + 8 * h;
      decb[(size_t)m * D_ + n0 + 16 * j + lm] = f2bf(tanhf(acc[j][r]));
    }
}

// ---------------- mel = dec @ mel_w + b (16x80 per wave) ----------------

__global__ void __launch_bounds__(256) k_mel(
    const __bf16* __restrict__ decb, const __bf16* __restrict__ melWT,
    const float* __restrict__ mel_b, float* __restrict__ mel_out,
    __bf16* __restrict__ x0) {
  int wave = (blockIdx.x * blockDim.x + threadIdx.x) >> 5;
  int mTile = wave;
  if (mTile * 16 >= ROWS_) return;
  int l = threadIdx.x & 31, lm = l & 15, h = l >> 4;
  v8f acc[5];
#pragma unroll
  for (int j = 0; j < 5; ++j)
#pragma unroll
    for (int r = 0; r < 8; ++r) acc[j][r] = mel_b[16 * j + lm];
  const __bf16* A = decb + (size_t)mTile * 16 * D_;
#pragma unroll
  for (int kc = 0; kc < D_ / 32; ++kc) {
    v16bf a = load_tile(A + kc * 32, D_);
#pragma unroll
    for (int j = 0; j < 5; ++j)
      acc[j] = wmma_bf16(a, load_tile(melWT + (size_t)(16 * j) * D_ + kc * 32, D_), acc[j]);
  }
#pragma unroll
  for (int j = 0; j < 5; ++j)
#pragma unroll
    for (int r = 0; r < 8; ++r) {
      int m = mTile * 16 + r + 8 * h;
      int nn = 16 * j + lm;
      mel_out[(size_t)m * MEL_ + nn] = acc[j][r];
      x0[(size_t)m * MELP_ + nn] = f2bf(acc[j][r]);
    }
}

// ---------------- postnet conv1d (SAME, K=5): NT n-tiles per wave, A reused NT x ----------------

template <int NT>
__global__ void __launch_bounds__(256) k_conv(
    const __bf16* __restrict__ x, int Cinp,
    const __bf16* __restrict__ wT, const float* __restrict__ bias, int Cout,
    __bf16* __restrict__ y, const float* __restrict__ mel, float* __restrict__ outf,
    int lastLayer) {
  int wave = (blockIdx.x * blockDim.x + threadIdx.x) >> 5;
  int nGroups = Cout / (16 * NT);
  int mTile = wave / nGroups, g = wave - mTile * nGroups;
  if (mTile * 16 >= ROWS_) return;
  int l = threadIdx.x & 31, lm = l & 15, h = l >> 4;
  int n0 = g * 16 * NT;
  v8f acc[NT];
#pragma unroll
  for (int j = 0; j < NT; ++j)
#pragma unroll
    for (int r = 0; r < 8; ++r) acc[j][r] = bias[n0 + 16 * j + lm];
  int m0 = mTile * 16;
  int bbase = m0 >> 12, tbase = m0 & (MM_ - 1);
#pragma unroll
  for (int tap = 0; tap < 5; ++tap) {
    int tl = tbase + lm + tap - 2;
    bool valid = (tl >= 0) && (tl < MM_);
    const __bf16* arow = x + ((size_t)bbase * MM_ + (valid ? tl : 0)) * Cinp;
    __builtin_prefetch(arow, 0, 0);   // global_prefetch_b8
    const __bf16* wk = wT + (size_t)tap * Cout * Cinp + (size_t)n0 * Cinp;
    for (int kc = 0; kc < Cinp / 32; ++kc) {
      v16bf a = load_tile_row(arow + kc * 32, valid);
#pragma unroll
      for (int j = 0; j < NT; ++j)
        acc[j] = wmma_bf16(a, load_tile(wk + (size_t)(16 * j) * Cinp + kc * 32, Cinp), acc[j]);
    }
  }
#pragma unroll
  for (int j = 0; j < NT; ++j)
#pragma unroll
    for (int r = 0; r < 8; ++r) {
      int m = m0 + r + 8 * h;
      int nn = n0 + 16 * j + lm;
      if (lastLayer) {
        size_t o = (size_t)m * MEL_ + nn;
        outf[o] = mel[o] + acc[j][r];
      } else {
        y[(size_t)m * D_ + nn] = f2bf(tanhf(acc[j][r]));
      }
    }
}

// ---------------- host ----------------

extern "C" void kernel_launch(void* const* d_in, const int* in_sizes, int n_in,
                              void* d_out, int out_size, void* d_ws, size_t ws_size,
                              hipStream_t stream) {
  if (n_in < 27) return;
  const int*   texts    = (const int*)d_in[0];
  const int*   src_lens = (const int*)d_in[1];
  const float* spk      = (const float*)d_in[3];
  const int*   d_tg     = (const int*)d_in[4];
  const float* emb      = (const float*)d_in[6];
  const float* enc_w    = (const float*)d_in[7];
  const float* enc_b    = (const float*)d_in[8];
  const float* pp_w1    = (const float*)d_in[9];
  const float* pp_b1    = (const float*)d_in[10];
  const float* pp_w2    = (const float*)d_in[11];
  const float* pp_b2    = (const float*)d_in[12];
  const float* dec_w    = (const float*)d_in[13];
  const float* dec_b    = (const float*)d_in[14];
  const float* mel_w    = (const float*)d_in[15];
  const float* mel_b    = (const float*)d_in[16];
  const float* cw[5]; const float* cb[5];
  for (int i = 0; i < 5; ++i) { cw[i] = (const float*)d_in[17 + 2 * i]; cb[i] = (const float*)d_in[18 + 2 * i]; }

  char* ws = (char*)d_ws; size_t off = 0;
  auto alloc = [&](size_t bytes) -> void* {
    void* p = ws + off; off = (off + bytes + 255) & ~(size_t)255; return p;
  };
  __bf16* embB   = (__bf16*)alloc((size_t)256 * D_ * 2);
  __bf16* encWT  = (__bf16*)alloc((size_t)D_ * D_ * 2);
  __bf16* WaT    = (__bf16*)alloc((size_t)D_ * D_ * 2);
  __bf16* WbT    = (__bf16*)alloc((size_t)D_ * D_ * 2);
  __bf16* WcT    = (__bf16*)alloc((size_t)D_ * D_ * 2);
  __bf16* W2T    = (__bf16*)alloc((size_t)D_ * D_ * 2);
  __bf16* decWT  = (__bf16*)alloc((size_t)D_ * D_ * 2);
  __bf16* melWT  = (__bf16*)alloc((size_t)MEL_ * D_ * 2);
  const int chans[6] = {MEL_, 256, 256, 256, 256, MEL_};
  __bf16* convT[5];
  int cinp[5];
  for (int i = 0; i < 5; ++i) {
    cinp[i] = (chans[i] == MEL_) ? MELP_ : chans[i];
    convT[i] = (__bf16*)alloc((size_t)5 * chans[i + 1] * cinp[i] * 2);
  }
  __bf16* h_si_bf   = (__bf16*)alloc((size_t)MTOK_ * D_ * 2);
  __bf16* h_sd_bf   = (__bf16*)alloc((size_t)MTOK_ * D_ * 2);
  float*  h_sd_f    = (float*)alloc((size_t)MTOK_ * D_ * 4);
  float*  pre       = (float*)alloc((size_t)MTOK_ * D_ * 4);
  __bf16* outrow_bf = (__bf16*)alloc((size_t)MTOK_ * D_ * 2);
  int*    cum       = (int*)alloc((size_t)B_ * T_ * 4);
  int*    total     = (int*)alloc(256);
  int*    idx       = (int*)alloc((size_t)B_ * MM_ * 4);
  __bf16* expb      = (__bf16*)alloc((size_t)ROWS_ * D_ * 2);
  __bf16* decb      = (__bf16*)alloc((size_t)ROWS_ * D_ * 2);
  __bf16* x0        = (__bf16*)alloc((size_t)ROWS_ * MELP_ * 2);
  __bf16* x1        = (__bf16*)alloc((size_t)ROWS_ * 256 * 2);
  __bf16* x2        = (__bf16*)alloc((size_t)ROWS_ * 256 * 2);

  float* mel_out  = (float*)d_out;
  float* post_out = mel_out + (size_t)ROWS_ * MEL_;

  auto TR = [&](const float* s, __bf16* d, int K, int N, int Kp) {
    int tot = N * Kp;
    k_transpose_bf16<<<(tot + 255) / 256, 256, 0, stream>>>(s, d, K, N, Kp);
  };

  // weight prep
  k_convert_bf16<<<(256 * D_ + 255) / 256, 256, 0, stream>>>(emb, embB, 256 * D_);
  TR(enc_w, encWT, D_, D_, D_);
  TR(pp_w1 + 0 * D_ * D_, WaT, D_, D_, D_);
  TR(pp_w1 + 1 * D_ * D_, WbT, D_, D_, D_);
  TR(pp_w1 + 2 * D_ * D_, WcT, D_, D_, D_);
  TR(pp_w2, W2T, D_, D_, D_);
  TR(dec_w, decWT, D_, D_, D_);
  TR(mel_w, melWT, D_, MEL_, D_);
  for (int i = 0; i < 5; ++i)
    for (int k = 0; k < 5; ++k)
      TR(cw[i] + (size_t)k * chans[i] * chans[i + 1],
         convT[i] + (size_t)k * chans[i + 1] * cinp[i],
         chans[i], chans[i + 1], cinp[i]);

  // encoder
  {
    int tiles = (MTOK_ / 16) * (D_ / 16);
    k_encoder<<<(tiles + 7) / 8, 256, 0, stream>>>(texts, src_lens, spk, embB, encWT, enc_b,
                                                   h_si_bf, h_sd_bf, h_sd_f);
  }
  // parallel prenet part
  {
    int tiles = (MTOK_ / 16) * (D_ / 16);
    k_pre<<<(tiles + 7) / 8, 256, 0, stream>>>(h_sd_bf, h_si_bf, WaT, WbT, pp_b1, pre);
  }
  // sequential scan: one persistent workgroup, 272 KB LDS, async weight preload
  {
    size_t shmem = (size_t)(2 * D_ * D_ + 2 * B_ * D_) * sizeof(__bf16);
    k_scan<<<1, 512, shmem, stream>>>(WcT, W2T, pre, pp_b2, h_sd_f, outrow_bf);
  }
  // length regulator
  k_cumsum<<<1, 32, 0, stream>>>(d_tg, src_lens, cum, total);
  k_index<<<(B_ * MM_ + 255) / 256, 256, 0, stream>>>(cum, idx);
  k_expand<<<(B_ * MM_ * D_ + 255) / 256, 256, 0, stream>>>(outrow_bf, idx, total, expb);
  // decoder: 16x64 per wave
  {
    int waves = (ROWS_ / 16) * (D_ / 64);
    k_dec<<<(waves + 7) / 8, 256, 0, stream>>>(expb, decWT, dec_b, decb);
  }
  // mel projection (+ padded bf16 copy for postnet): 16x80 per wave
  k_zero_bf16<<<(ROWS_ * MELP_ + 255) / 256, 256, 0, stream>>>(x0, ROWS_ * MELP_);
  {
    int waves = ROWS_ / 16;
    k_mel<<<(waves + 7) / 8, 256, 0, stream>>>(decb, melWT, mel_b, mel_out, x0);
  }
  // postnet
  const __bf16* cx[5] = {x0, x1, x2, x1, x2};
  __bf16* cy[5] = {x1, x2, x1, x2, nullptr};
  for (int i = 0; i < 5; ++i) {
    int cout = chans[i + 1];
    if (cout == 256) {
      int waves = (ROWS_ / 16) * (cout / 64);
      k_conv<4><<<(waves + 7) / 8, 256, 0, stream>>>(cx[i], cinp[i], convT[i], cb[i], cout,
                                                     cy[i], mel_out, post_out, 0);
    } else { // cout == 80, NT=5, one n-group
      int waves = ROWS_ / 16;
      k_conv<5><<<(waves + 7) / 8, 256, 0, stream>>>(cx[i], cinp[i], convT[i], cb[i], cout,
                                                     cy[i], mel_out, post_out, 1);
    }
  }
}
